// TelechatAttention_87574383166020
// MI455X (gfx1250) — matrix operand order repaired and checked by reading, not compiled, not run on hardware
//
#include <hip/hip_runtime.h>
#include <hip/hip_bf16.h>
#include <math.h>

#define NUM_HEADS 32
#define HEAD_DIM  160
#define HIDDEN    (NUM_HEADS * HEAD_DIM)   // 5120
#define SEQ       2048
#define ROPE_BASE 10000.0f

typedef __attribute__((ext_vector_type(16))) __bf16 v16bf;
typedef __attribute__((ext_vector_type(8)))  __bf16 v8bf;
typedef __attribute__((ext_vector_type(8)))  float  v8f;

union FragBF { v16bf v; v8bf h[2]; };

// A-matrix 16x32 bf16 fragment (ISA 7.12.2): lane m(<16) holds K 0..7,16..23;
// lane m+16 holds K 8..15,24..31.  rowp = &A[row][k0]
__device__ __forceinline__ v16bf load_a_frag(const __bf16* rowp, int hk) {
    FragBF f;
    const __bf16* p = rowp + 8 * hk;
    f.h[0] = *(const v8bf*)(p);
    f.h[1] = *(const v8bf*)(p + 16);
    return f.v;
}
// B-matrix 32x16 bf16 fragment: lanes 0-15 hold K=0..7,16..23 of col n, lanes 16-31
// hold K=8..15,24..31.  rowp = &W[n][k0]  (W stored [N][K] so B[k][n] = W[n][k])
__device__ __forceinline__ v16bf load_b_frag(const __bf16* rowp, int hk) {
    FragBF f;
    const __bf16* p = rowp + 8 * hk;
    f.h[0] = *(const v8bf*)(p);
    f.h[1] = *(const v8bf*)(p + 16);
    return f.v;
}

#define WMMA_BF16(A, B, C) \
    __builtin_amdgcn_wmma_f32_16x16x32_bf16(false, (A), false, (B), (short)0, (C), false, false)

// CDNA5 async VMEM->LDS copy (ASYNCcnt-tracked); builtin not probe-confirmed,
// so use inline asm per CDNA5 ISA. GV mode: vdst = per-lane
// LDS byte offset, vaddr = per-lane 64-bit global address.
#define ASYNC_LDS_B128(ldsoff, gaddr)                                        \
    asm volatile("global_load_async_to_lds_b128 %0, %1, off"                 \
                 :: "v"(ldsoff), "v"(gaddr) : "memory")
#define WAIT_ASYNCCNT0() asm volatile("s_wait_asynccnt 0" ::: "memory")

// ---------------- elementwise conversion f32 -> bf16 (vectorized x4) --------
__global__ void cvt_f32_to_bf16(const float* __restrict__ src,
                                __bf16* __restrict__ dst, long long n) {
    long long i = ((long long)blockIdx.x * blockDim.x + threadIdx.x) * 4;
    if (i + 3 < n) {
        float4 f = *(const float4*)(src + i);
        dst[i + 0] = (__bf16)f.x;
        dst[i + 1] = (__bf16)f.y;
        dst[i + 2] = (__bf16)f.z;
        dst[i + 3] = (__bf16)f.w;
    }
}

// ---------------- RoPE tables: cos/sin [SEQ][HEAD_DIM] fp32 -----------------
__global__ void rope_tables(float* __restrict__ cosT, float* __restrict__ sinT) {
    int idx = blockIdx.x * blockDim.x + threadIdx.x;
    if (idx >= SEQ * HEAD_DIM) return;
    int s = idx / HEAD_DIM, j = idx % HEAD_DIM;
    int i = j % (HEAD_DIM / 2);
    float inv = powf(ROPE_BASE, -(2.0f * (float)i) / (float)HEAD_DIM);
    float ang = (float)s * inv;
    cosT[idx] = cosf(ang);
    sinT[idx] = sinf(ang);
}

// ------ RoPE on q (fp32 [S][H]) -> bf16 head-major q_h[h][s][d] -------------
__global__ void rope_q_scatter(const float* __restrict__ q,
                               const float* __restrict__ cosT,
                               const float* __restrict__ sinT,
                               __bf16* __restrict__ qh) {
    int idx = blockIdx.x * blockDim.x + threadIdx.x;
    if (idx >= SEQ * HIDDEN) return;
    int s = idx / HIDDEN, c = idx % HIDDEN;
    int h = c / HEAD_DIM, j = c % HEAD_DIM;
    const float* row = q + (long long)s * HIDDEN + h * HEAD_DIM;
    float x = row[j];
    float other = (j < HEAD_DIM / 2) ? -row[j + HEAD_DIM / 2] : row[j - HEAD_DIM / 2];
    float r = x * cosT[s * HEAD_DIM + j] + other * sinT[s * HEAD_DIM + j];
    qh[((long long)h * SEQ + s) * HEAD_DIM + j] = (__bf16)r;
}

// ------ RoPE on k + transpose v from kv (fp32 [S][2H]) ----------------------
__global__ void rope_kv_scatter(const float* __restrict__ kv,
                                const float* __restrict__ cosT,
                                const float* __restrict__ sinT,
                                __bf16* __restrict__ kh,
                                __bf16* __restrict__ vt) {
    int idx = blockIdx.x * blockDim.x + threadIdx.x;
    if (idx >= SEQ * HIDDEN) return;
    int s = idx / HIDDEN, c = idx % HIDDEN;
    int h = c / HEAD_DIM, j = c % HEAD_DIM;
    const float* base = kv + (long long)s * (2 * HIDDEN) + h * (2 * HEAD_DIM);
    float kx = base[j];
    float other = (j < HEAD_DIM / 2) ? -base[j + HEAD_DIM / 2] : base[j - HEAD_DIM / 2];
    float kr = kx * cosT[s * HEAD_DIM + j] + other * sinT[s * HEAD_DIM + j];
    kh[((long long)h * SEQ + s) * HEAD_DIM + j] = (__bf16)kr;
    vt[((long long)h * HEAD_DIM + j) * SEQ + s] = (__bf16)base[HEAD_DIM + j];
}

// ---------------- generic bf16 WMMA GEMM:  C[M][N] = A[M][K] @ W[N][K]^T ----
// block = 256 thr = 8 waves in 2x4; wave tile 32x64; block tile 64x256.
__global__ void __launch_bounds__(256)
gemm_bf16_wmma(const __bf16* __restrict__ A, const __bf16* __restrict__ W,
               float* __restrict__ C, int M, int N, int K,
               const float* __restrict__ bias, const float* __restrict__ residual) {
    int lane = threadIdx.x & 31, wave = threadIdx.x >> 5;
    int wm = wave & 1, wn = wave >> 1;
    int m0 = blockIdx.y * 64 + wm * 32;
    int n0 = blockIdx.x * 256 + wn * 64;
    int nl = lane & 15, hk = lane >> 4;

    v8f acc[2][4] = {};
    const __bf16* arow0 = A + (long long)(m0 + nl) * K;
    const __bf16* arow1 = A + (long long)(m0 + 16 + nl) * K;
    const __bf16* wrow[4];
#pragma unroll
    for (int ni = 0; ni < 4; ni++)
        wrow[ni] = W + (long long)(n0 + ni * 16 + nl) * K;

    for (int k0 = 0; k0 < K; k0 += 32) {
        v16bf a0 = load_a_frag(arow0 + k0, hk);
        v16bf a1 = load_a_frag(arow1 + k0, hk);
#pragma unroll
        for (int ni = 0; ni < 4; ni++) {
            v16bf b = load_b_frag(wrow[ni] + k0, hk);
            acc[0][ni] = WMMA_BF16(a0, b, acc[0][ni]);
            acc[1][ni] = WMMA_BF16(a1, b, acc[1][ni]);
        }
    }
    // C/D layout: VGPR r, lanes 0-15 -> M=r, lanes 16-31 -> M=r+8; N = lane%16
#pragma unroll
    for (int mi = 0; mi < 2; mi++)
#pragma unroll
        for (int ni = 0; ni < 4; ni++)
#pragma unroll
            for (int r = 0; r < 8; r++) {
                int row = m0 + mi * 16 + r + 8 * hk;
                int col = n0 + ni * 16 + nl;
                float v = acc[mi][ni][r];
                if (bias)     v += bias[col];
                if (residual) v += residual[(long long)row * N + col];
                C[(long long)row * N + col] = v;
            }
}

// ---------------- flash-style causal attention ------------------------------
// grid (SEQ/128, NUM_HEADS); 8 waves; each wave owns 16 q rows.
// K/V blocks (32 keys) are staged once per workgroup into LDS with CDNA5
// async VMEM->LDS copies; all waves compute WMMA fragments from LDS.
__global__ void __launch_bounds__(256)
attn_wmma(const __bf16* __restrict__ qh_all, const __bf16* __restrict__ kh_all,
          const __bf16* __restrict__ vt_all, __bf16* __restrict__ ctx_out) {
    __shared__ __bf16 kbuf[32 * HEAD_DIM];       // K block: [key][d]   10 KB
    __shared__ __bf16 vbuf[HEAD_DIM * 32];       // V block: [d][key]   10 KB
    __shared__ __bf16 plds[8 * 16 * 32];         // per-wave P tiles     8 KB

    int lane = threadIdx.x & 31, wave = threadIdx.x >> 5;
    int nl = lane & 15, hk = lane >> 4;
    int h = blockIdx.y;
    int qbase = blockIdx.x * 128 + wave * 16;
    int kb_end = blockIdx.x * 128 + 128;         // causal bound for this block

    const __bf16* qh  = qh_all + (long long)h * SEQ * HEAD_DIM;
    const __bf16* khp = kh_all + (long long)h * SEQ * HEAD_DIM;
    const __bf16* vt  = vt_all + (long long)h * HEAD_DIM * SEQ;
    const float inv_norm = 0.079056941504f;      // 1/sqrt(160)

    // q rows live in registers for the whole kernel (5 x 16x32 A-fragments)
    v16bf qfrag[5];
    {
        const __bf16* qrow = qh + (long long)(qbase + nl) * HEAD_DIM;
#pragma unroll
        for (int dk = 0; dk < 5; dk++) qfrag[dk] = load_a_frag(qrow + dk * 32, hk);
    }

    v8f ctx[10];
#pragma unroll
    for (int dn = 0; dn < 10; dn++) ctx[dn] = (v8f)0.0f;
    float mrow[8], lrow[8];
#pragma unroll
    for (int r = 0; r < 8; r++) { mrow[r] = -1e30f; lrow[r] = 0.0f; }

    __bf16* pw = plds + wave * (16 * 32);

    for (int kb = 0; kb < kb_end; kb += 32) {
        // ---- cooperative async staging of K(32x160) and V(160x32) ----------
        __syncthreads();                          // prior block's readers done
#pragma unroll
        for (int it = 0; it < 5; it++) {          // 1280 16B chunks / 256 thr
            int c = threadIdx.x + it * 256;
            unsigned ldsoff;
            unsigned long long gaddr;
            if (c < 640) {                        // K: row r, 20 chunks/row
                int r = c / 20, part = c % 20;
                gaddr  = (unsigned long long)(uintptr_t)
                         (khp + (long long)(kb + r) * HEAD_DIM + part * 8);
                ldsoff = (unsigned)(uintptr_t)&kbuf[r * HEAD_DIM + part * 8];
            } else {                              // V: d row, 4 chunks/row
                int cc = c - 640, d = cc / 4, part = cc % 4;
                gaddr  = (unsigned long long)(uintptr_t)
                         (vt + (long long)d * SEQ + kb + part * 8);
                ldsoff = (unsigned)(uintptr_t)&vbuf[d * 32 + part * 8];
            }
            ASYNC_LDS_B128(ldsoff, gaddr);
        }
        WAIT_ASYNCCNT0();
        __syncthreads();                          // staged data visible to all

        if (kb <= qbase + 15) {                   // wave-uniform causal skip
            // ---- S = q @ k^T for 16x32 block (K-dim = 160), from LDS ------
            v8f sacc[2] = {};
            const __bf16* krow0 = kbuf + nl * HEAD_DIM;
            const __bf16* krow1 = kbuf + (16 + nl) * HEAD_DIM;
#pragma unroll
            for (int dk = 0; dk < 5; dk++) {
                v16bf b0 = load_b_frag(krow0 + dk * 32, hk);
                v16bf b1 = load_b_frag(krow1 + dk * 32, hk);
                sacc[0] = WMMA_BF16(qfrag[dk], b0, sacc[0]);
                sacc[1] = WMMA_BF16(qfrag[dk], b1, sacc[1]);
            }
            // ---- scale + causal mask ----
            float ps[2][8];
#pragma unroll
            for (int ni = 0; ni < 2; ni++)
#pragma unroll
                for (int r = 0; r < 8; r++) {
                    int col = kb + ni * 16 + nl;
                    int row = qbase + r + 8 * hk;
                    float sv = sacc[ni][r] * inv_norm;
                    ps[ni][r] = (col > row) ? -1e30f : sv;
                }
            // ---- online softmax (rows per (r,half); cols across 16 lanes) --
#pragma unroll
            for (int r = 0; r < 8; r++) {
                float mx = fmaxf(ps[0][r], ps[1][r]);
                for (int off = 1; off < 16; off <<= 1)
                    mx = fmaxf(mx, __shfl_xor(mx, off, 32));
                float mnew = fmaxf(mrow[r], mx);
                float corr = __expf(mrow[r] - mnew);
                float p0 = __expf(ps[0][r] - mnew);
                float p1 = __expf(ps[1][r] - mnew);
                float rs = p0 + p1;
                for (int off = 1; off < 16; off <<= 1)
                    rs += __shfl_xor(rs, off, 32);
                lrow[r] = lrow[r] * corr + rs;
                mrow[r] = mnew;
#pragma unroll
                for (int dn = 0; dn < 10; dn++) ctx[dn][r] *= corr;
                // stage P (C-layout) into LDS for re-read in A-frag layout
                pw[(r + 8 * hk) * 32 + nl]      = (__bf16)p0;
                pw[(r + 8 * hk) * 32 + 16 + nl] = (__bf16)p1;
            }
            // ---- ctx += P(16x32) @ V(32x160); V tile in LDS, [d][key] -----
            v16bf pa = load_a_frag(pw + nl * 32, hk);
#pragma unroll
            for (int dn = 0; dn < 10; dn++) {
                v16bf bv = load_b_frag(vbuf + (dn * 16 + nl) * 32, hk);
                ctx[dn] = WMMA_BF16(pa, bv, ctx[dn]);
            }
        }
    }
    // ---- normalize + write ctx bf16 as [S][HIDDEN] for the output GEMM ----
#pragma unroll
    for (int r = 0; r < 8; r++) {
        float inv = 1.0f / lrow[r];
        int row = qbase + r + 8 * hk;
#pragma unroll
        for (int dn = 0; dn < 10; dn++) {
            int col = h * HEAD_DIM + dn * 16 + nl;
            ctx_out[(long long)row * HIDDEN + col] = (__bf16)(ctx[dn][r] * inv);
        }
    }
}

// ---------------------------------------------------------------------------
extern "C" void kernel_launch(void* const* d_in, const int* in_sizes, int n_in,
                              void* d_out, int out_size, void* d_ws, size_t ws_size,
                              hipStream_t stream) {
    (void)in_sizes; (void)n_in; (void)out_size; (void)ws_size;
    const float* hidden   = (const float*)d_in[0];
    const float* residual = (const float*)d_in[1];
    /* d_in[2] = attention_mask (causal, statically known) */
    const float* Wq  = (const float*)d_in[3];
    const float* Wkv = (const float*)d_in[4];
    const float* Wd  = (const float*)d_in[5];
    const float* bd  = (const float*)d_in[6];
    float* out = (float*)d_out;

    const long long SZ_X   = (long long)SEQ * HIDDEN;       // 10,485,760
    const long long SZ_W   = (long long)HIDDEN * HIDDEN;    // 26,214,400
    const long long SZ_WKV = 2 * SZ_W;

    char* ws = (char*)d_ws;
    size_t off = 0;
    auto alloc = [&](size_t bytes) -> void* {
        void* p = ws + off;
        off += (bytes + 255) & ~(size_t)255;
        return p;
    };
    __bf16* x_bf   = (__bf16*)alloc(SZ_X   * 2);
    __bf16* wq_bf  = (__bf16*)alloc(SZ_W   * 2);
    __bf16* wkv_bf = (__bf16*)alloc(SZ_WKV * 2);
    __bf16* wd_bf  = (__bf16*)alloc(SZ_W   * 2);
    float*  q_f    = (float*) alloc(SZ_X   * 4);
    float*  kv_f   = (float*) alloc(SZ_X * 2 * 4);
    __bf16* qh     = (__bf16*)alloc(SZ_X * 2);
    __bf16* khb    = (__bf16*)alloc(SZ_X * 2);
    __bf16* vtb    = (__bf16*)alloc(SZ_X * 2);
    __bf16* ctxb   = (__bf16*)alloc(SZ_X * 2);
    float*  cosT   = (float*) alloc((size_t)SEQ * HEAD_DIM * 4);
    float*  sinT   = (float*) alloc((size_t)SEQ * HEAD_DIM * 4);

    const int T = 256;
    auto blk4 = [](long long n) { return (unsigned)((n / 4 + 255) / 256); };

    // 1) fp32 -> bf16 conversions (weights become L2-resident bf16)
    cvt_f32_to_bf16<<<blk4(SZ_X),   T, 0, stream>>>(hidden, x_bf,   SZ_X);
    cvt_f32_to_bf16<<<blk4(SZ_W),   T, 0, stream>>>(Wq,     wq_bf,  SZ_W);
    cvt_f32_to_bf16<<<blk4(SZ_WKV), T, 0, stream>>>(Wkv,    wkv_bf, SZ_WKV);
    cvt_f32_to_bf16<<<blk4(SZ_W),   T, 0, stream>>>(Wd,     wd_bf,  SZ_W);

    // 2) RoPE tables
    rope_tables<<<(SEQ * HEAD_DIM + T - 1) / T, T, 0, stream>>>(cosT, sinT);

    // 3) Q and KV projections (bf16 WMMA, fp32 accum)
    gemm_bf16_wmma<<<dim3(HIDDEN / 256, SEQ / 64), T, 0, stream>>>(
        x_bf, wq_bf, q_f, SEQ, HIDDEN, HIDDEN, nullptr, nullptr);
    gemm_bf16_wmma<<<dim3(2 * HIDDEN / 256, SEQ / 64), T, 0, stream>>>(
        x_bf, wkv_bf, kv_f, SEQ, 2 * HIDDEN, HIDDEN, nullptr, nullptr);

    // 4) RoPE + head-major scatter (v transposed)
    rope_q_scatter<<<(unsigned)((SZ_X + T - 1) / T), T, 0, stream>>>(q_f, cosT, sinT, qh);
    rope_kv_scatter<<<(unsigned)((SZ_X + T - 1) / T), T, 0, stream>>>(kv_f, cosT, sinT, khb, vtb);

    // 5) causal flash attention (async-LDS staged K/V)
    attn_wmma<<<dim3(SEQ / 128, NUM_HEADS), T, 0, stream>>>(qh, khb, vtb, ctxb);

    // 6) output projection + bias + residual -> fp32 d_out
    gemm_bf16_wmma<<<dim3(HIDDEN / 256, SEQ / 64), T, 0, stream>>>(
        ctxb, wd_bf, out, SEQ, HIDDEN, HIDDEN, bd, residual);
}
